// NonLocalBlock2D_with_mask_Res_13322988552681
// MI455X (gfx1250) — matrix operand order repaired and checked by reading, not compiled, not run on hardware
//
#include <hip/hip_runtime.h>
#include <hip/hip_bf16.h>

typedef __attribute__((ext_vector_type(16))) _Float16 v16h;
typedef __attribute__((ext_vector_type(8)))  _Float16 v8h;
typedef __attribute__((ext_vector_type(8)))  float    v8f;

#define WMMA(a, b, c) \
  __builtin_amdgcn_wmma_f32_16x16x32_f16(false, (a), false, (b), (short)0, (c), false, false)

#define NBATCH 2
#define NPIX   4096   // H*W
#define CC     128
#define NTOK   (NBATCH * NPIX)

// ---------- WMMA fragment loaders (contiguous k order: element s holds k = 16*(lane/16) + s) ----------

// A (16x32 f16, MxK): lanes 0-15 row M=lane&15, elements 0-15 = K[kb+ho..kb+ho+15]
// ho = 0 for lanes 0-15, 16 for lanes 16-31. Two 16-byte vector loads.
__device__ __forceinline__ v16h load_a_row(const _Float16* rowp, int kbase, int lane) {
  int ho = (lane >> 4) << 4;
  const _Float16* p = rowp + kbase + ho;
  v8h lo = *(const v8h*)(p);
  v8h hi = *(const v8h*)(p + 8);
  return __builtin_shufflevector(lo, hi, 0, 1, 2, 3, 4, 5, 6, 7, 8, 9, 10, 11, 12, 13, 14, 15);
}

__device__ __forceinline__ v16h load_a_frag(const _Float16* base, int ld, int kbase, int lane) {
  return load_a_row(base + (size_t)(lane & 15) * ld, kbase, lane);
}

// B (32x16 f16, KxN) with B[k][n] = M[n][k], M row-major (leading dim ld):
// lane holds column N=ncol+(lane&15); elements h = K[kbase + (lane>=16?16:0) + h].
// Contiguous 32 bytes per lane -> one v16h load (2x b128).
__device__ __forceinline__ v16h load_b_tn(const _Float16* m, int ld, int kbase, int ncol, int lane) {
  int col = ncol + (lane & 15);
  int ko  = kbase + ((lane >> 4) << 4);
  return *(const v16h*)(m + (size_t)col * ld + ko);
}

// ---------- prep kernels ----------

__global__ void cvt_f32_f16(const float* __restrict__ a, _Float16* __restrict__ o, int n) {
  int i = blockIdx.x * 256 + threadIdx.x;
  if (i < n) o[i] = (_Float16)a[i];
}

// convert + transpose a batch of 128x128 matrices: out[m][c][r] = in[m][r][c]
__global__ void cvt_t128(const float* __restrict__ in, _Float16* __restrict__ out, int nmat) {
  size_t i = (size_t)blockIdx.x * 256 + threadIdx.x;
  if (i >= (size_t)nmat * 16384) return;
  size_t m = i >> 14;
  int rc = (int)(i & 16383);
  int r = rc >> 7, c = rc & 127;
  out[(m << 14) + ((size_t)c << 7) + r] = (_Float16)in[i];
}

__global__ void mask_prep(const float* __restrict__ mask, float* __restrict__ m) {
  int i = blockIdx.x * 256 + threadIdx.x;
  if (i >= NTOK) return;
  float t = mask[i];
  float r = (t > 0.f) ? 0.f : t;   // zero positively-masked (resize is identity here)
  r = 1.f - r;
  m[i] = r * (1.f - t);            // * inverted mask
}

// ---------- 1x1 projections: g/theta/phi = x @ W + b (three GEMMs, WMMA) ----------
// Weights are pre-transposed (wT[cout][cin]).  g is stored transposed: gT[b][c][key].

__global__ __launch_bounds__(256) void proj_wmma(
    const _Float16* __restrict__ xh,
    const _Float16* __restrict__ wgT, const _Float16* __restrict__ wtT, const _Float16* __restrict__ wpT,
    const float* __restrict__ bg, const float* __restrict__ bt, const float* __restrict__ bp,
    _Float16* __restrict__ gT, _Float16* __restrict__ thq, _Float16* __restrict__ phq) {
  int wave = threadIdx.x >> 5, lane = threadIdx.x & 31;
  int job  = blockIdx.x * 8 + wave;            // 3 * 512 * 8 = 12288 jobs
  int nt   = job & 7;
  int rest = job >> 3;
  int mt   = rest & 511;
  int wsel = rest >> 9;
  const _Float16* w   = (wsel == 0) ? wgT : (wsel == 1) ? wtT : wpT;
  const float*    bia = (wsel == 0) ? bg  : (wsel == 1) ? bt  : bp;

  const _Float16* abase = xh + (size_t)mt * 16 * CC;
  v8f c = {};
#pragma unroll
  for (int s = 0; s < 4; ++s) {
    v16h a  = load_a_frag(abase, CC, s * 32, lane);
    v16h bf = load_b_tn(w, CC, s * 32, nt * 16, lane);
    c = WMMA(a, bf, c);
  }
  int lc = lane & 15, hf = lane >> 4;
  int col = nt * 16 + lc;
  float bb = bia[col];
#pragma unroll
  for (int i = 0; i < 8; ++i) {
    size_t row = (size_t)mt * 16 + i + hf * 8;      // global token
    _Float16 v = (_Float16)(c[i] + bb);
    if (wsel == 0) {
      size_t b   = row >> 12;
      size_t key = row & (NPIX - 1);
      gT[b * (size_t)CC * NPIX + (size_t)col * NPIX + key] = v;
    } else if (wsel == 1) {
      thq[row * CC + col] = v;
    } else {
      phq[row * CC + col] = v;
    }
  }
}

// ---------- fused flash attention (mask-gated columns) + y@ww epilogue ----------

__global__ __launch_bounds__(256) void attn_wmma(
    const _Float16* __restrict__ theta, const _Float16* __restrict__ phi, const _Float16* __restrict__ gT,
    const float* __restrict__ mgate, const _Float16* __restrict__ wwT, const float* __restrict__ bw,
    float* __restrict__ wy, _Float16* __restrict__ act) {
  __shared__ _Float16 slds[8][16 * CC];        // per-wave staging (P: 16x32, y: 16x128)
  int wave = threadIdx.x >> 5, lane = threadIdx.x & 31;
  int job = blockIdx.x * 8 + wave;             // 512 jobs: (batch, qtile)
  int b  = job >> 8;
  int qt = job & 255;
  int lc = lane & 15, hf = lane >> 4;

  const _Float16* th_b = theta + (size_t)b * NPIX * CC;
  const _Float16* ph_b = phi   + (size_t)b * NPIX * CC;
  const _Float16* g_b  = gT    + (size_t)b * CC * NPIX;   // gT[c][key]
  const float*    m_b  = mgate + (size_t)b * NPIX;
  _Float16* pl = slds[wave];

  // theta A-fragments for the 16 query rows (K = 128 in 4 slices), kept resident
  v16h ta[4];
#pragma unroll
  for (int s = 0; s < 4; ++s) ta[s] = load_a_frag(th_b + (size_t)qt * 16 * CC, CC, s * 32, lane);

  float M[8], L[8];
  v8f acc[8];
#pragma unroll
  for (int i = 0; i < 8; ++i) { M[i] = -1e30f; L[i] = 0.f; acc[i] = (v8f){}; }

  for (int kt = 0; kt < 128; ++kt) {           // 32 keys per iteration
    int k0 = kt * 32;
    v8f s0 = {}, s1 = {};
#pragma unroll
    for (int s = 0; s < 4; ++s) {
      s0 = WMMA(ta[s], load_b_tn(ph_b, CC, s * 32, k0,      lane), s0);
      s1 = WMMA(ta[s], load_b_tn(ph_b, CC, s * 32, k0 + 16, lane), s1);
    }
    float mc0 = m_b[k0 + lc], mc1 = m_b[k0 + 16 + lc];

#pragma unroll
    for (int i = 0; i < 8; ++i) {
      float rmax = fmaxf(s0[i], s1[i]);
#pragma unroll
      for (int off = 8; off >= 1; off >>= 1) rmax = fmaxf(rmax, __shfl_xor(rmax, off, 16));
      float newM  = fmaxf(M[i], rmax);
      float alpha = __expf(M[i] - newM);
      float p0 = __expf(s0[i] - newM);
      float p1 = __expf(s1[i] - newM);
      float rs = p0 + p1;                      // un-gated: softmax denominator
#pragma unroll
      for (int off = 8; off >= 1; off >>= 1) rs += __shfl_xor(rs, off, 16);
      L[i] = L[i] * alpha + rs;
      M[i] = newM;
#pragma unroll
      for (int n = 0; n < 8; ++n) acc[n][i] *= alpha;
      int row = i + hf * 8;                    // stage gated P (16x32) for transpose
      pl[row * 32 + lc]      = (_Float16)(p0 * mc0);
      pl[row * 32 + 16 + lc] = (_Float16)(p1 * mc1);
    }
    asm volatile("s_wait_dscnt 0" ::: "memory");
    v16h pa = load_a_frag(pl, 32, 0, lane);    // P as A-matrix (16x32)
#pragma unroll
    for (int n = 0; n < 8; ++n)
      acc[n] = WMMA(pa, load_b_tn(g_b, NPIX, k0, n * 16, lane), acc[n]);
  }

  // y = acc / L  -> stage f16 (16 x 128)
#pragma unroll
  for (int n = 0; n < 8; ++n)
#pragma unroll
    for (int i = 0; i < 8; ++i)
      pl[(i + hf * 8) * CC + n * 16 + lc] = (_Float16)(acc[n][i] / L[i]);
  asm volatile("s_wait_dscnt 0" ::: "memory");

  // w_y = y @ ww + bw (WMMA epilogue)
  v16h ya[4];
#pragma unroll
  for (int s = 0; s < 4; ++s) ya[s] = load_a_frag(pl, CC, s * 32, lane);
#pragma unroll
  for (int n = 0; n < 8; ++n) {
    v8f c = {};
#pragma unroll
    for (int s = 0; s < 4; ++s) c = WMMA(ya[s], load_b_tn(wwT, CC, s * 32, n * 16, lane), c);
    int col = n * 16 + lc;
    float bb = bw[col];
#pragma unroll
    for (int i = 0; i < 8; ++i) {
      size_t row = (size_t)b * NPIX + qt * 16 + i + hf * 8;
      float v = c[i] + bb;
      wy[row * CC + col]  = v;
      act[row * CC + col] = (_Float16)v;
    }
  }
}

// ---------- 3x3 reflect-pad conv as implicit GEMM (WMMA) ----------

__device__ __forceinline__ int reflect64(int v) {
  v = (v < 0) ? -v : v;
  return (v > 63) ? (126 - v) : v;
}

__global__ __launch_bounds__(256) void conv3x3_wmma(
    const _Float16* __restrict__ in, const _Float16* __restrict__ wt9T,  // 9 x [cout][cin] f16
    const float* __restrict__ bias, float* __restrict__ out) {
  int wave = threadIdx.x >> 5, lane = threadIdx.x & 31;
  int job = blockIdx.x * 8 + wave;             // 4096 jobs
  int ct = job & 7;
  int pt = job >> 3;                           // 0..511
  int b    = pt >> 8;
  int tpix = pt & 255;
  int h  = tpix >> 2;
  int w0 = (tpix & 3) << 4;
  int lc = lane & 15, hf = lane >> 4;

  v8f c = {};
#pragma unroll
  for (int dh = 0; dh < 3; ++dh) {
    int ih = reflect64(h + dh - 1);
#pragma unroll
    for (int dw = 0; dw < 3; ++dw) {
      int iw = reflect64(w0 + lc + dw - 1);    // per-lane pixel row of A
      const _Float16* arow = in + ((size_t)b * NPIX + ih * 64 + iw) * CC;
      const _Float16* wt   = wt9T + (size_t)(dh * 3 + dw) * CC * CC;
#pragma unroll
      for (int s = 0; s < 4; ++s) {
        v16h a  = load_a_row(arow, s * 32, lane);
        v16h bf = load_b_tn(wt, CC, s * 32, ct * 16, lane);
        c = WMMA(a, bf, c);
      }
    }
  }
  int col = ct * 16 + lc;
  float bb = bias[col];
#pragma unroll
  for (int i = 0; i < 8; ++i) {
    int pw = w0 + i + hf * 8;
    out[((size_t)b * NPIX + h * 64 + pw) * CC + col] = c[i] + bb;
  }
}

// ---------- instance norm: per-(b,c) stats then normalize ----------

__global__ void inorm_stats(const float* __restrict__ xin, float* __restrict__ mean, float* __restrict__ var) {
  int bc = blockIdx.x;                          // 256 = B*C
  int b = bc >> 7, ch = bc & 127;
  const float* p = xin + (size_t)b * NPIX * CC + ch;
  float s = 0.f, ss = 0.f;
  for (int i = threadIdx.x; i < NPIX; i += 256) {
    float v = p[(size_t)i * CC];
    s += v; ss += v * v;
  }
  __shared__ float sh[256], sh2[256];
  sh[threadIdx.x] = s; sh2[threadIdx.x] = ss;
  __syncthreads();
  for (int o = 128; o > 0; o >>= 1) {
    if (threadIdx.x < o) { sh[threadIdx.x] += sh[threadIdx.x + o]; sh2[threadIdx.x] += sh2[threadIdx.x + o]; }
    __syncthreads();
  }
  if (threadIdx.x == 0) {
    float m = sh[0] * (1.f / NPIX);
    mean[bc] = m;
    var[bc]  = sh2[0] * (1.f / NPIX) - m * m;
  }
}

__global__ void norm_relu(const float* __restrict__ xin, const float* __restrict__ mean,
                          const float* __restrict__ var, const float* __restrict__ gamma,
                          const float* __restrict__ beta, _Float16* __restrict__ out) {
  size_t i = (size_t)blockIdx.x * 256 + threadIdx.x;   // NTOK*CC = 2^20
  int c = (int)(i & 127);
  int b = (int)(i >> 19);
  int bc = b * CC + c;
  float v = (xin[i] - mean[bc]) * rsqrtf(var[bc] + 1e-3f) * gamma[c] + beta[c];
  out[i] = (_Float16)fmaxf(v, 0.f);
}

__global__ void norm_add(const float* __restrict__ xin, const float* __restrict__ mean,
                         const float* __restrict__ var, const float* __restrict__ gamma,
                         const float* __restrict__ beta, float* __restrict__ wy,
                         _Float16* __restrict__ act) {
  size_t i = (size_t)blockIdx.x * 256 + threadIdx.x;
  int c = (int)(i & 127);
  int b = (int)(i >> 19);
  int bc = b * CC + c;
  float hv = (xin[i] - mean[bc]) * rsqrtf(var[bc] + 1e-3f) * gamma[c] + beta[c];
  float nv = wy[i] + hv;
  wy[i]  = nv;
  act[i] = (_Float16)nv;
}

__global__ void blend(const float* __restrict__ x, const float* __restrict__ m,
                      const float* __restrict__ wy, float* __restrict__ z) {
  size_t i = (size_t)blockIdx.x * 256 + threadIdx.x;
  float mm = m[i >> 7];
  z[i] = mm * x[i] + (1.f - mm) * wy[i];
}

// ---------- host ----------

extern "C" void kernel_launch(void* const* d_in, const int* in_sizes, int n_in,
                              void* d_out, int out_size, void* d_ws, size_t ws_size,
                              hipStream_t stream) {
  const float* x       = (const float*)d_in[0];
  const float* mask    = (const float*)d_in[1];
  const float* wg      = (const float*)d_in[2];
  const float* bg      = (const float*)d_in[3];
  const float* wt      = (const float*)d_in[4];
  const float* bt      = (const float*)d_in[5];
  const float* wp      = (const float*)d_in[6];
  const float* bp      = (const float*)d_in[7];
  const float* ww      = (const float*)d_in[8];
  const float* bw      = (const float*)d_in[9];
  const float* res_w1  = (const float*)d_in[10];
  const float* res_b1  = (const float*)d_in[11];
  const float* res_g1  = (const float*)d_in[12];
  const float* res_be1 = (const float*)d_in[13];
  const float* res_w2  = (const float*)d_in[14];
  const float* res_b2  = (const float*)d_in[15];
  const float* res_g2  = (const float*)d_in[16];
  const float* res_be2 = (const float*)d_in[17];

  const size_t NE   = (size_t)NTOK * CC;   // 1,048,576
  const size_t WE   = (size_t)CC * CC;     // 16,384
  const size_t CWE  = 27 * WE;             // 442,368 per res weight set
  char* ws = (char*)d_ws;
  size_t off = 0;
  auto take = [&](size_t bytes) { char* p = ws + off; off += (bytes + 255) & ~(size_t)255; return p; };

  _Float16* xh  = (_Float16*)take(NE * 2);
  _Float16* wgT = (_Float16*)take(WE * 2);
  _Float16* wtT = (_Float16*)take(WE * 2);
  _Float16* wpT = (_Float16*)take(WE * 2);
  _Float16* wwT = (_Float16*)take(WE * 2);
  _Float16* w1T = (_Float16*)take(CWE * 2);
  _Float16* w2T = (_Float16*)take(CWE * 2);
  _Float16* gT  = (_Float16*)take(NE * 2);
  _Float16* thq = (_Float16*)take(NE * 2);
  _Float16* phq = (_Float16*)take(NE * 2);
  float*    mg  = (float*)take((size_t)NTOK * 4);
  float*    wy  = (float*)take(NE * 4);
  _Float16* act = (_Float16*)take(NE * 2);
  float*    tmp = (float*)take(NE * 4);
  _Float16* hb  = (_Float16*)take(NE * 2);
  float*    mn  = (float*)take(256 * 4);
  float*    vr  = (float*)take(256 * 4);

  // f16 conversions / weight transposes
  cvt_f32_f16<<<(int)((NE + 255) / 256), 256, 0, stream>>>(x, xh, (int)NE);
  cvt_t128<<<64, 256, 0, stream>>>(wg, wgT, 1);
  cvt_t128<<<64, 256, 0, stream>>>(wt, wtT, 1);
  cvt_t128<<<64, 256, 0, stream>>>(wp, wpT, 1);
  cvt_t128<<<64, 256, 0, stream>>>(ww, wwT, 1);
  cvt_t128<<<(int)((CWE + 255) / 256), 256, 0, stream>>>(res_w1, w1T, 27);
  cvt_t128<<<(int)((CWE + 255) / 256), 256, 0, stream>>>(res_w2, w2T, 27);
  mask_prep<<<NTOK / 256, 256, 0, stream>>>(mask, mg);

  // projections (g stored transposed)
  proj_wmma<<<1536, 256, 0, stream>>>(xh, wgT, wtT, wpT, bg, bt, bp, gT, thq, phq);

  // fused flash attention + w_y = y@ww
  attn_wmma<<<64, 256, 0, stream>>>(thq, phq, gT, mg, wwT, bw, wy, act);

  // 3 residual blocks
  for (int i = 0; i < 3; ++i) {
    conv3x3_wmma<<<512, 256, 0, stream>>>(act, w1T + (size_t)i * 9 * WE, res_b1 + i * CC, tmp);
    inorm_stats<<<256, 256, 0, stream>>>(tmp, mn, vr);
    norm_relu<<<(int)(NE / 256), 256, 0, stream>>>(tmp, mn, vr, res_g1 + i * CC, res_be1 + i * CC, hb);
    conv3x3_wmma<<<512, 256, 0, stream>>>(hb, w2T + (size_t)i * 9 * WE, res_b2 + i * CC, tmp);
    inorm_stats<<<256, 256, 0, stream>>>(tmp, mn, vr);
    norm_add<<<(int)(NE / 256), 256, 0, stream>>>(tmp, mn, vr, res_g2 + i * CC, res_be2 + i * CC, wy, act);
  }

  // final mask-gated blend
  blend<<<(int)(NE / 256), 256, 0, stream>>>(x, mg, wy, (float*)d_out);
}
